// PerPointVFModel_74062416053314
// MI455X (gfx1250) — matrix operand order repaired and checked by reading, not compiled, not run on hardware
//
#include <hip/hip_runtime.h>
#include <hip/hip_bf16.h>
#include <math.h>

typedef __attribute__((ext_vector_type(16))) __bf16 v16bf;
typedef __attribute__((ext_vector_type(8)))  float  v8f;

union BF16Op { v16bf v; unsigned int w[8]; unsigned short us[16]; };

__device__ __forceinline__ unsigned short f2bf(float f){
  unsigned int u = __builtin_bit_cast(unsigned int, f);
  u += 0x7FFFu + ((u >> 16) & 1u);          // round-to-nearest-even
  return (unsigned short)(u >> 16);
}
__device__ __forceinline__ float gelu_exact(float x){
  return 0.5f * x * (1.0f + erff(x * 0.7071067811865475f));
}

// ---------------- workspace layout (bytes) ----------------
#define OFF_QPAD  ((size_t)0)          // ushort bf16 [64][128]  (queries padded, pre-scaled 1/sqrt(128))
#define OFF_KW    ((size_t)16384)      // ushort [128][1024]
#define OFF_VW    ((size_t)278528)     // ushort [128][1024]
#define OFF_OW    ((size_t)540672)     // ushort [1024][128]
#define OFF_HW1   ((size_t)802816)     // ushort [256][2048]
#define OFF_HW2   ((size_t)1851392)    // ushort [128][256]
#define OFF_PRIOR ((size_t)1916928)    // float  [52][196]
#define OFF_KEYS  ((size_t)1957888)    // ushort [256][208][128]
#define OFF_VALST ((size_t)15589376)   // ushort [256][128][224]  (transposed vals, K-padded)
#define OFF_ATT   ((size_t)30269440)   // ushort [256][64][128]

__constant__ unsigned char c_mask[72] = {
  0,0,0,1,1,1,1,0,0,
  0,0,1,1,1,1,1,1,0,
  0,1,1,1,1,1,1,1,1,
  1,1,1,1,1,1,1,0,1,
  1,1,1,1,1,1,1,0,1,
  0,1,1,1,1,1,1,1,1,
  0,0,1,1,1,1,1,1,0,
  0,0,0,1,1,1,1,0,0
};

// ---------------- K0: weight conversion + prior + padding ----------------
__global__ void k_prep(const float* __restrict__ queries, const float* __restrict__ key_w,
                       const float* __restrict__ val_w, const float* __restrict__ out_w,
                       const float* __restrict__ hw1, const float* __restrict__ hw2,
                       char* __restrict__ ws){
  size_t idx = (size_t)blockIdx.x * 256 + threadIdx.x;
  if (idx < 8192){                               // queries padded to 64 rows, /sqrt(128)
    int q = (int)(idx >> 7), a = (int)(idx & 127);
    float v = (q < 52) ? queries[q*128 + a] * 0.08838834764831843f : 0.0f;
    ((unsigned short*)(ws + OFF_QPAD))[idx] = f2bf(v);
    return;
  }
  idx -= 8192;
  if (idx < 131072){ ((unsigned short*)(ws + OFF_KW))[idx] = f2bf(key_w[idx]); return; }
  idx -= 131072;
  if (idx < 131072){ ((unsigned short*)(ws + OFF_VW))[idx] = f2bf(val_w[idx]); return; }
  idx -= 131072;
  if (idx < 131072){ ((unsigned short*)(ws + OFF_OW))[idx] = f2bf(out_w[idx]); return; }
  idx -= 131072;
  if (idx < 524288){ ((unsigned short*)(ws + OFF_HW1))[idx] = f2bf(hw1[idx]); return; }
  idx -= 524288;
  if (idx < 32768){ ((unsigned short*)(ws + OFF_HW2))[idx] = f2bf(hw2[idx]); return; }
  idx -= 32768;
  if (idx < 10192){                              // ATTN_PRIOR
    int q = (int)(idx / 196), p = (int)(idx % 196);
    int cnt = -1, cr = 0, cc = 0;
    for (int r = 0; r < 8 && cnt < q; r++)
      for (int c = 0; c < 9 && cnt < q; c++)
        if (c_mask[r*9 + c]){ cnt++; if (cnt == q){ cr = 13 - (r*13)/7; cc = 13 - (c*13)/8; } }
    int dr = p/14 - cr, dc = p%14 - cc;
    ((float*)(ws + OFF_PRIOR))[idx] = -(float)(dr*dr + dc*dc) / 18.0f;
    return;
  }
  idx -= 10192;
  if (idx < 524288){                             // zero valsT pad rows p=208..223
    int b = (int)(idx / 2048); int rem = (int)(idx % 2048);
    int a = rem >> 4; int pp = 208 + (rem & 15);
    ((unsigned short*)(ws + OFF_VALST))[(size_t)b*128*224 + a*224 + pp] = 0;
  }
}

// ---------------- K1: keys & vals (fused, latent read once, K-chunk 64) ----------------
__global__ void __launch_bounds__(256) k_kv(const float* __restrict__ latent,
                                            const float* __restrict__ patch_pos,
                                            char* __restrict__ ws){
  __shared__ unsigned short s_a[16*64];     // 2 KB : patch tile (bf16), K-chunk 64
  __shared__ unsigned int   s_bk[128*32];   // 16 KB: key_w tile
  __shared__ unsigned int   s_bv[128*32];   // 16 KB: val_w tile
  const int tid = threadIdx.x, wave = tid >> 5, lane = tid & 31;
  const int h = (lane >> 4) & 1, ln16 = lane & 15;
  const int pt = blockIdx.x, b = blockIdx.y;
  const unsigned int* kw = (const unsigned int*)(ws + OFF_KW);
  const unsigned int* vw = (const unsigned int*)(ws + OFF_VW);
  unsigned short* keys  = (unsigned short*)(ws + OFF_KEYS)  + (size_t)b*208*128;
  unsigned short* valsT = (unsigned short*)(ws + OFF_VALST) + (size_t)b*128*224;
  const float* patch = latent + ((size_t)b*197 + 1)*1024;
  const int nbase = wave * 16;              // 8 waves x 16 cols = 128 cols
  v8f ak = {}, av = {};
  for (int kk = 0; kk < 1024; kk += 64){
    // stage A: 16 rows x 64 k (256 float4 groups, 1 per thread)
    {
      int row = tid >> 4, k4 = (tid & 15) * 4, p = pt*16 + row;
      float4 f = (p < 196) ? *(const float4*)&patch[(size_t)p*1024 + kk + k4]
                           : make_float4(0.f, 0.f, 0.f, 0.f);
      unsigned int lo = (unsigned int)f2bf(f.x) | ((unsigned int)f2bf(f.y) << 16);
      unsigned int hi = (unsigned int)f2bf(f.z) | ((unsigned int)f2bf(f.w) << 16);
      *(uint2*)&s_a[row*64 + k4] = make_uint2(lo, hi);
    }
    // stage B: 64 rows(a) x 32 uints each (512 uint4 groups per matrix)
    for (int g = tid; g < 512; g += 256){
      int a = g >> 3, jp4 = (g & 7) * 4;
      *(uint4*)&s_bk[a*32 + jp4] = *(const uint4*)&kw[a*512 + (kk >> 1) + jp4];
      *(uint4*)&s_bv[a*32 + jp4] = *(const uint4*)&vw[a*512 + (kk >> 1) + jp4];
    }
    __syncthreads();
    #pragma unroll
    for (int s = 0; s < 64; s += 32){
      BF16Op A, Bk, Bv;
      #pragma unroll
      for (int i = 0; i < 8; i++){
        int kp = h*16 + i*2;
        A.w[i] = *(const unsigned int*)&s_a[ln16*64 + s + kp];
      }
      #pragma unroll
      for (int i = 0; i < 8; i++){
        int j = (s >> 1) + h*8 + i;
        Bk.w[i] = s_bk[(nbase + ln16)*32 + j];
        Bv.w[i] = s_bv[(nbase + ln16)*32 + j];
      }
      ak = __builtin_amdgcn_wmma_f32_16x16x32_bf16(false, A.v, false, Bk.v, (short)0, ak, false, false);
      av = __builtin_amdgcn_wmma_f32_16x16x32_bf16(false, A.v, false, Bv.v, (short)0, av, false, false);
    }
    __syncthreads();
  }
  const int a0 = nbase + ln16;
  // keys: column-strided scalar stores (+patch_pos), rows p = pt*16 + h*8 + r
  #pragma unroll
  for (int r = 0; r < 8; r++){
    int p = pt*16 + h*8 + r;
    float kr = ak[r] + ((p < 196) ? patch_pos[p*128 + a0] : 0.0f);
    keys[p*128 + a0] = f2bf(kr);
  }
  // valsT: 8 consecutive p per lane -> one 16B store
  {
    unsigned int w0 = (unsigned int)f2bf(av[0]) | ((unsigned int)f2bf(av[1]) << 16);
    unsigned int w1 = (unsigned int)f2bf(av[2]) | ((unsigned int)f2bf(av[3]) << 16);
    unsigned int w2 = (unsigned int)f2bf(av[4]) | ((unsigned int)f2bf(av[5]) << 16);
    unsigned int w3 = (unsigned int)f2bf(av[6]) | ((unsigned int)f2bf(av[7]) << 16);
    *(uint4*)&valsT[a0*224 + pt*16 + h*8] = make_uint4(w0, w1, w2, w3);
  }
}

// ---------------- K2: attention (logits + prior + softmax + attn@V) ----------------
__global__ void __launch_bounds__(256) k_attn(char* __restrict__ ws){
  extern __shared__ __align__(16) char smem[];
  float*          L      = (float*)smem;                       // [64][208]
  unsigned short* s_attn = (unsigned short*)(smem + 64*208*4); // [64][224]
  const int tid = threadIdx.x, wave = tid >> 5, lane = tid & 31;
  const int h = (lane >> 4) & 1, ln16 = lane & 15;
  const int b = blockIdx.x;
  const unsigned int* qpad  = (const unsigned int*)(ws + OFF_QPAD);
  const float*        prior = (const float*)(ws + OFF_PRIOR);
  const unsigned int* keys  = (const unsigned int*)(ws + OFF_KEYS)  + (size_t)b*208*64;
  const unsigned int* valsT = (const unsigned int*)(ws + OFF_VALST) + (size_t)b*128*112;
  unsigned short*     att   = (unsigned short*)(ws + OFF_ATT)       + (size_t)b*64*128;

  for (int ti = wave; ti < 52; ti += 8){                // 4x13 logit tiles
    int mt = ti / 13, ptile = ti % 13;
    v8f acc = {};
    for (int kk = 0; kk < 128; kk += 32){
      BF16Op A, B;
      #pragma unroll
      for (int i = 0; i < 8; i++){
        int kp = h*16 + i*2;
        A.w[i] = qpad[(mt*16 + ln16)*64 + ((kk + kp) >> 1)];
      }
      int p = ptile*16 + ln16;
      #pragma unroll
      for (int i = 0; i < 8; i++)
        B.w[i] = keys[p*64 + ((kk + h*16) >> 1) + i];
      acc = __builtin_amdgcn_wmma_f32_16x16x32_bf16(false, A.v, false, B.v, (short)0, acc, false, false);
    }
    #pragma unroll
    for (int r = 0; r < 8; r++)
      L[(mt*16 + h*8 + r)*208 + ptile*16 + ln16] = acc[r];
  }
  __syncthreads();
  for (int idx = tid; idx < 64*208; idx += 256){        // prior + mask
    int q = idx / 208, p = idx % 208;
    float v = L[idx];
    if (p >= 196) v = -1e30f;
    else if (q < 52) v += prior[q*196 + p];
    L[idx] = v;
  }
  __syncthreads();
  if (tid < 64){                                        // row softmax -> bf16 attn
    int q = tid;
    float mx = -1e30f;
    for (int p = 0; p < 196; p++) mx = fmaxf(mx, L[q*208 + p]);
    float s = 0.0f;
    for (int p = 0; p < 196; p++) s += expf(L[q*208 + p] - mx);
    float inv = 1.0f / s;
    for (int p = 0; p < 196; p++) s_attn[q*224 + p] = f2bf(expf(L[q*208 + p] - mx) * inv);
    for (int p = 196; p < 224; p++) s_attn[q*224 + p] = 0;
  }
  __syncthreads();
  for (int ti = wave; ti < 32; ti += 8){                // attended = attn @ vals
    int mt = ti >> 3, nt = ti & 7;
    v8f acc = {};
    for (int kk = 0; kk < 224; kk += 32){
      BF16Op A, B;
      #pragma unroll
      for (int i = 0; i < 8; i++){
        int kp = h*16 + i*2;
        A.w[i] = *(const unsigned int*)&s_attn[(mt*16 + ln16)*224 + kk + kp];
      }
      int a = nt*16 + ln16;
      #pragma unroll
      for (int i = 0; i < 8; i++)
        B.w[i] = valsT[a*112 + ((kk + h*16) >> 1) + i];
      acc = __builtin_amdgcn_wmma_f32_16x16x32_bf16(false, A.v, false, B.v, (short)0, acc, false, false);
    }
    #pragma unroll
    for (int r = 0; r < 8; r++)
      att[(mt*16 + h*8 + r)*128 + nt*16 + ln16] = f2bf(acc[r]);
  }
}

// ---------------- K3: out_w proj + MLP head + correction ----------------
__global__ void __launch_bounds__(256) k_mlp(const float* __restrict__ latent,
    const float* __restrict__ out_b, const float* __restrict__ hw1f,
    const float* __restrict__ hb1, const float* __restrict__ hg1, const float* __restrict__ hbe1,
    const float* __restrict__ hb2, const float* __restrict__ hg2, const float* __restrict__ hbe2,
    const float* __restrict__ hw3, const float* __restrict__ hb3,
    const float* __restrict__ rw1, const float* __restrict__ rb1, const float* __restrict__ rg,
    const float* __restrict__ rbe, const float* __restrict__ rw2, const float* __restrict__ rb2,
    const float* __restrict__ alpha, char* __restrict__ ws, float* __restrict__ out){
  extern __shared__ __align__(16) char smem[];
  unsigned short* s_pf  = (unsigned short*)smem;             // [64][1024] bf16
  float*          s_h1  = (float*)(smem + 131072);           // [64][256]
  unsigned short* s_h1b = (unsigned short*)(smem + 196608);  // [64][256] bf16
  float*          s_h2  = (float*)(smem + 229376);           // [64][128]
  float*          s_cls = (float*)(smem + 262144);           // [256]
  float*          s_pred= (float*)(smem + 263168);           // [64]
  float*          s_x   = (float*)(smem + 263424);           // [104]
  float*          s_stat= (float*)(smem + 263936);           // [2]
  const int tid = threadIdx.x, wave = tid >> 5, lane = tid & 31;
  const int h = (lane >> 4) & 1, ln16 = lane & 15;
  const int b = blockIdx.x;
  const float* cls = latent + (size_t)b*197*1024;

  // cls half of hw1 (tiny VALU dot: 0.5 MF per block)
  for (int j = tid; j < 256; j += 256){
    float acc = 0.0f;
    for (int d = 0; d < 1024; d++) acc += cls[d] * hw1f[j*2048 + 1024 + d];
    s_cls[j] = acc;
  }
  // pf = attended @ out_w^T  (M=64, K=128, N=1024)
  const unsigned int* attu = (const unsigned int*)(ws + OFF_ATT) + (size_t)b*64*64;
  const unsigned int* owu  = (const unsigned int*)(ws + OFF_OW);
  for (int ti = wave; ti < 256; ti += 8){
    int mt = ti >> 6, nt = ti & 63;
    v8f acc = {};
    for (int kk = 0; kk < 128; kk += 32){
      BF16Op A, B;
      #pragma unroll
      for (int i = 0; i < 8; i++){
        int kp = h*16 + i*2;
        A.w[i] = attu[(mt*16 + ln16)*64 + ((kk + kp) >> 1)];
      }
      int d = nt*16 + ln16;
      #pragma unroll
      for (int i = 0; i < 8; i++)
        B.w[i] = owu[d*64 + ((kk + h*16) >> 1) + i];
      acc = __builtin_amdgcn_wmma_f32_16x16x32_bf16(false, A.v, false, B.v, (short)0, acc, false, false);
    }
    int d = nt*16 + ln16;
    #pragma unroll
    for (int r = 0; r < 8; r++)
      s_pf[(mt*16 + h*8 + r)*1024 + d] = f2bf(acc[r] + out_b[d]);
  }
  __syncthreads();
  // h1 = pf @ hw1_left^T + hb1 + cls_part  (M=64, K=1024, N=256)
  const unsigned int* hw1u = (const unsigned int*)(ws + OFF_HW1);
  for (int ti = wave; ti < 64; ti += 8){
    int mt = ti >> 4, nt = ti & 15;
    v8f acc = {};
    for (int kk = 0; kk < 1024; kk += 32){
      BF16Op A, B;
      #pragma unroll
      for (int i = 0; i < 8; i++){
        int kp = h*16 + i*2;
        A.w[i] = *(const unsigned int*)&s_pf[(mt*16 + ln16)*1024 + kk + kp];
      }
      int j = nt*16 + ln16;
      #pragma unroll
      for (int i = 0; i < 8; i++)
        B.w[i] = hw1u[j*1024 + ((kk + h*16) >> 1) + i];
      acc = __builtin_amdgcn_wmma_f32_16x16x32_bf16(false, A.v, false, B.v, (short)0, acc, false, false);
    }
    int j = nt*16 + ln16;
    #pragma unroll
    for (int r = 0; r < 8; r++)
      s_h1[(mt*16 + h*8 + r)*256 + j] = acc[r] + hb1[j] + s_cls[j];
  }
  __syncthreads();
  if (tid < 64){                                        // LN(256) + GELU
    int q = tid;
    float m = 0.0f;
    for (int j = 0; j < 256; j++) m += s_h1[q*256 + j];
    m *= (1.0f/256.0f);
    float v = 0.0f;
    for (int j = 0; j < 256; j++){ float d = s_h1[q*256 + j] - m; v += d*d; }
    v *= (1.0f/256.0f);
    float rs = rsqrtf(v + 1e-5f);
    for (int j = 0; j < 256; j++){
      float y = (s_h1[q*256 + j] - m)*rs*hg1[j] + hbe1[j];
      s_h1b[q*256 + j] = f2bf(gelu_exact(y));
    }
  }
  __syncthreads();
  // h2 = h1 @ hw2^T + hb2  (M=64, K=256, N=128)
  const unsigned int* hw2u = (const unsigned int*)(ws + OFF_HW2);
  for (int ti = wave; ti < 32; ti += 8){
    int mt = ti >> 3, nt = ti & 7;
    v8f acc = {};
    for (int kk = 0; kk < 256; kk += 32){
      BF16Op A, B;
      #pragma unroll
      for (int i = 0; i < 8; i++){
        int kp = h*16 + i*2;
        A.w[i] = *(const unsigned int*)&s_h1b[(mt*16 + ln16)*256 + kk + kp];
      }
      int j = nt*16 + ln16;
      #pragma unroll
      for (int i = 0; i < 8; i++)
        B.w[i] = hw2u[j*128 + ((kk + h*16) >> 1) + i];
      acc = __builtin_amdgcn_wmma_f32_16x16x32_bf16(false, A.v, false, B.v, (short)0, acc, false, false);
    }
    int j = nt*16 + ln16;
    #pragma unroll
    for (int r = 0; r < 8; r++)
      s_h2[(mt*16 + h*8 + r)*128 + j] = acc[r] + hb2[j];
  }
  __syncthreads();
  if (tid < 64){                                        // LN(128) + GELU (in place)
    int q = tid;
    float m = 0.0f;
    for (int j = 0; j < 128; j++) m += s_h2[q*128 + j];
    m *= (1.0f/128.0f);
    float v = 0.0f;
    for (int j = 0; j < 128; j++){ float d = s_h2[q*128 + j] - m; v += d*d; }
    v *= (1.0f/128.0f);
    float rs = rsqrtf(v + 1e-5f);
    for (int j = 0; j < 128; j++){
      float y = (s_h2[q*128 + j] - m)*rs*hg2[j] + hbe2[j];
      s_h2[q*128 + j] = gelu_exact(y);
    }
  }
  __syncthreads();
  if (tid < 52){                                        // head
    int q = tid;
    float acc = hb3[0];
    for (int a = 0; a < 128; a++) acc += s_h2[q*128 + a] * hw3[a];
    s_pred[q] = acc;
  }
  __syncthreads();
  if (tid < 104){                                       // correction branch
    int t = tid;
    float x = rb1[t];
    for (int q = 0; q < 52; q++) x += s_pred[q] * rw1[t*52 + q];
    s_x[t] = x;
  }
  __syncthreads();
  if (tid == 0){
    float m = 0.0f;
    for (int t = 0; t < 104; t++) m += s_x[t];
    m *= (1.0f/104.0f);
    float v = 0.0f;
    for (int t = 0; t < 104; t++){ float d = s_x[t] - m; v += d*d; }
    v *= (1.0f/104.0f);
    s_stat[0] = m; s_stat[1] = rsqrtf(v + 1e-5f);
  }
  __syncthreads();
  if (tid < 104){
    int t = tid;
    float y = (s_x[t] - s_stat[0]) * s_stat[1] * rg[t] + rbe[t];
    s_x[t] = gelu_exact(y);
  }
  __syncthreads();
  if (tid < 52){
    int q = tid;
    float corr = rb2[q];
    for (int t = 0; t < 104; t++) corr += s_x[t] * rw2[q*104 + t];
    float sg = 1.0f / (1.0f + expf(-alpha[0]));
    float p = s_pred[q] + sg * corr;
    p = (p < 0.1f) ? 0.0f : p;
    p = fminf(fmaxf(p, 0.0f), 35.0f);
    out[b*52 + q] = p;
  }
}

extern "C" void kernel_launch(void* const* d_in, const int* in_sizes, int n_in,
                              void* d_out, int out_size, void* d_ws, size_t ws_size,
                              hipStream_t stream){
  (void)in_sizes; (void)n_in; (void)out_size; (void)ws_size;
  const float* latent    = (const float*)d_in[0];
  const float* queries   = (const float*)d_in[1];
  const float* patch_pos = (const float*)d_in[2];
  const float* key_w     = (const float*)d_in[3];
  const float* val_w     = (const float*)d_in[4];
  const float* out_w     = (const float*)d_in[5];
  const float* out_b     = (const float*)d_in[6];
  const float* hw1       = (const float*)d_in[7];
  const float* hb1       = (const float*)d_in[8];
  const float* hg1       = (const float*)d_in[9];
  const float* hbe1      = (const float*)d_in[10];
  const float* hw2       = (const float*)d_in[11];
  const float* hb2       = (const float*)d_in[12];
  const float* hg2       = (const float*)d_in[13];
  const float* hbe2      = (const float*)d_in[14];
  const float* hw3       = (const float*)d_in[15];
  const float* hb3       = (const float*)d_in[16];
  const float* rw1       = (const float*)d_in[17];
  const float* rb1       = (const float*)d_in[18];
  const float* rg        = (const float*)d_in[19];
  const float* rbe       = (const float*)d_in[20];
  const float* rw2       = (const float*)d_in[21];
  const float* rb2       = (const float*)d_in[22];
  const float* alpha     = (const float*)d_in[23];
  char* ws = (char*)d_ws;
  float* out = (float*)d_out;

  // total prep elements = 1,492,944
  k_prep<<<5833, 256, 0, stream>>>(queries, key_w, val_w, out_w, hw1, hw2, ws);
  k_kv<<<dim3(13, 256), 256, 0, stream>>>(latent, patch_pos, ws);
  k_attn<<<256, 256, 64*208*4 + 64*224*2, stream>>>(ws);
  k_mlp<<<256, 256, 264192, stream>>>(latent, out_b, hw1, hb1, hg1, hbe1,
                                      hb2, hg2, hbe2, hw3, hb3,
                                      rw1, rb1, rg, rbe, rw2, rb2, alpha, ws, out);
}
